// MultiheadSelfAttention_6614249636106
// MI455X (gfx1250) — matrix operand order repaired and checked by reading, not compiled, not run on hardware
//
#include <hip/hip_runtime.h>
#include <hip/hip_bf16.h>

// ---------------------------------------------------------------------------
// MI455X (gfx1250) multi-head self-attention, bf16 WMMA everywhere.
// Pipeline: cvt(f32->bf16) -> 3x GEMM (QKV proj, double-buffered TDM tiles) ->
//           RoPE/reorder -> flash-attention (TDM K-tile, WMMA QK^T + PV) ->
//           GEMM (out proj, f32 out)
// ---------------------------------------------------------------------------

#define D_MODEL 1024
#define NHEADS  16
#define DK      64
#define BATCH   2
#define SEQ     2048
#define MTOT    (BATCH * SEQ)      // 4096

typedef __attribute__((ext_vector_type(16))) __bf16 v16bf;
typedef __attribute__((ext_vector_type(8)))  float  v8f;
typedef __attribute__((ext_vector_type(4)))  unsigned int u32x4;
typedef __attribute__((ext_vector_type(8)))  int    i32x8;
typedef __attribute__((ext_vector_type(4)))  int    i32x4;

union Frag {               // one WMMA bf16 A/B operand: 16 bf16 per lane
    uint4 u4[2];
    v16bf v;
};

__device__ __forceinline__ v8f wmma_bf16(const Frag& a, const Frag& b, v8f c) {
    // D = A(16x32 bf16) * B(32x16 bf16) + C(16x16 f32)
    return __builtin_amdgcn_wmma_f32_16x16x32_bf16(
        /*neg_a=*/false, a.v, /*neg_b=*/false, b.v,
        /*c_mod=*/(short)0, c, /*reuse_a=*/false, /*reuse_b=*/false);
}

// ---------------------------------------------------------------------------
// Tensor Data Mover: 2D bf16 tile (tile_d1 rows x tile_d0 elems) from a
// row-major matrix (row stride = stride_elems) into LDS at lds_off, with
// HW row padding: after every (2 << pad_int_code) DWORDs of data, skip
// (pad_amt_code + 1) DWORDs in LDS. D# packing per CDNA5 ISA sec. 8.3/8.4,
// groups 2/3 zero (2-D tensor). Tracked by TENSORcnt.
// ---------------------------------------------------------------------------
__device__ __forceinline__ void tdm_load_tile_2d(unsigned lds_off,
                                                 const void* gaddr,
                                                 unsigned tensor_d0,
                                                 unsigned tensor_d1,
                                                 unsigned tile_d0,
                                                 unsigned tile_d1,
                                                 unsigned stride_elems,
                                                 unsigned pad_int_code,
                                                 unsigned pad_amt_code) {
    unsigned long long ga = (unsigned long long)gaddr;
    u32x4 g0;
    g0[0] = 1u;                                        // count=1, user D#
    g0[1] = lds_off;                                   // lds_addr
    g0[2] = (unsigned)(ga & 0xffffffffu);              // global_addr[31:0]
    g0[3] = (unsigned)((ga >> 32) & 0x01ffffffu)       // global_addr[56:32]
            | (2u << 30);                              // type=2 ("image")
    i32x8 g1;
    g1[0] = (int)((1u << 16)                 // data_size=1 -> 2 bytes
                | (1u << 20)                 // pad_enable
                | (pad_int_code << 22)       // pad_interval
                | (pad_amt_code << 25));     // pad_amount
    g1[1] = (int)((tensor_d0 & 0xffffu) << 16);        // tensor_dim0[15:0]
    g1[2] = (int)(((tensor_d0 >> 16) & 0xffffu)        // tensor_dim0[31:16]
                | ((tensor_d1 & 0xffffu) << 16));      // tensor_dim1[15:0]
    g1[3] = (int)(((tensor_d1 >> 16) & 0xffffu)        // tensor_dim1[31:16]
                | ((tile_d0 & 0xffffu) << 16));        // tile_dim0
    g1[4] = (int)(tile_d1 & 0xffffu);                  // tile_dim1 (tile_dim2=0)
    g1[5] = (int)stride_elems;                         // tensor_dim0_stride[31:0]
    g1[6] = 0;                                         // stride[47:32], d1_stride lo
    g1[7] = 0;
    i32x4 z4 = {0, 0, 0, 0};
#if defined(__clang_major__) && __clang_major__ >= 23
    i32x8 z8 = {0, 0, 0, 0, 0, 0, 0, 0};
    __builtin_amdgcn_tensor_load_to_lds(g0, g1, z4, z4, z8, 0);
#else
    __builtin_amdgcn_tensor_load_to_lds(g0, g1, z4, z4, 0);
#endif
}

// ---------------------------------------------------------------------------
// 1) fp32 -> bf16 streaming convert
// ---------------------------------------------------------------------------
__global__ void cvt_f32_bf16(const float* __restrict__ src,
                             __bf16* __restrict__ dst, int n) {
    int i = blockIdx.x * blockDim.x + threadIdx.x;
    if (i < n) dst[i] = (__bf16)src[i];
}

// ---------------------------------------------------------------------------
// 2) Tiled WMMA GEMM:  C[M,N] = A[M,K] * Bt[N,K]^T   (contraction on fast dim)
//    Block tile 128x128, BK=32. 8 waves: 4 (M) x 2 (N), each wave 32x64.
//    Double-buffered: TDM DMA of tile kt+1 overlaps WMMA compute on tile kt.
// ---------------------------------------------------------------------------
template <bool OUT_BF16>
__global__ __launch_bounds__(256) void gemm_wmma(const __bf16* __restrict__ A,
                                                 const __bf16* __restrict__ Bt,
                                                 void* __restrict__ Cout,
                                                 int M, int N, int K) {
    __shared__ __bf16 Al[2][128 * 40];   // rows padded to 40 (80B, 16B aligned)
    __shared__ __bf16 Bl[2][128 * 40];

    const int tid  = threadIdx.x;
    const int lane = tid & 31;
    const int w    = tid >> 5;
    const int wm   = w & 3;           // wave M index (0..3)
    const int wn   = w >> 2;          // wave N index (0..1)
    const int lhi  = lane >> 4;       // half-wave select
    const int llo  = lane & 15;
    const int m0   = blockIdx.y * 128;
    const int n0   = blockIdx.x * 128;

    const unsigned al_off[2] = {(unsigned)(size_t)(void*)&Al[0][0],
                                (unsigned)(size_t)(void*)&Al[1][0]};
    const unsigned bl_off[2] = {(unsigned)(size_t)(void*)&Bl[0][0],
                                (unsigned)(size_t)(void*)&Bl[1][0]};

    v8f zero = {};
    v8f acc[2][4];
#pragma unroll
    for (int mi = 0; mi < 2; ++mi)
#pragma unroll
        for (int ni = 0; ni < 4; ++ni) acc[mi][ni] = zero;

    // Prologue: DMA first K-tile into buffer 0.
    if (w == 0) {
        tdm_load_tile_2d(al_off[0], &A[(size_t)m0 * K], K, 128, 32, 128, K, 3, 3);
        tdm_load_tile_2d(bl_off[0], &Bt[(size_t)n0 * K], K, 128, 32, 128, K, 3, 3);
    }

    int cur = 0;
    for (int kt = 0; kt < K; kt += 32) {
        if (w == 0) __builtin_amdgcn_s_wait_tensorcnt(0);  // buf[cur] ready
        __syncthreads();
        if (w == 0 && kt + 32 < K) {    // kick off next tile into buf[cur^1]
            tdm_load_tile_2d(al_off[cur ^ 1], &A[(size_t)m0 * K + kt + 32],
                             K, 128, 32, 128, K, 3, 3);
            tdm_load_tile_2d(bl_off[cur ^ 1], &Bt[(size_t)n0 * K + kt + 32],
                             K, 128, 32, 128, K, 3, 3);
        }

        // ---- fragments (ISA 7.12.2 layouts) ----
        const __bf16* Ac = Al[cur];
        const __bf16* Bc = Bl[cur];
        Frag af[2], bfr[4];
#pragma unroll
        for (int mi = 0; mi < 2; ++mi) {
            const __bf16* p = &Ac[(wm * 32 + mi * 16 + llo) * 40 + lhi * 8];
            af[mi].u4[0] = *(const uint4*)p;          // K = base .. base+7
            af[mi].u4[1] = *(const uint4*)(p + 16);   // K = 16+base ..
        }
#pragma unroll
        for (int ni = 0; ni < 4; ++ni) {
            const __bf16* p = &Bc[(wn * 64 + ni * 16 + llo) * 40 + lhi * 8];
            bfr[ni].u4[0] = *(const uint4*)p;         // K = 8*lhi .. +7
            bfr[ni].u4[1] = *(const uint4*)(p + 16);  // K = 16+8*lhi .. +7
        }
#pragma unroll
        for (int mi = 0; mi < 2; ++mi)
#pragma unroll
            for (int ni = 0; ni < 4; ++ni)
                acc[mi][ni] = wmma_bf16(af[mi], bfr[ni], acc[mi][ni]);
        __syncthreads();                 // buf[cur] free for reuse
        cur ^= 1;
    }

    // ---- epilogue: C/D layout M = r + 8*lhi, N = llo ----
#pragma unroll
    for (int mi = 0; mi < 2; ++mi)
#pragma unroll
        for (int ni = 0; ni < 4; ++ni) {
            int n = n0 + wn * 64 + ni * 16 + llo;
#pragma unroll
            for (int r = 0; r < 8; ++r) {
                int m = m0 + wm * 32 + mi * 16 + r + lhi * 8;
                float v = acc[mi][ni][r];
                if (OUT_BF16)
                    ((__bf16*)Cout)[(size_t)m * N + n] = (__bf16)v;
                else
                    ((float*)Cout)[(size_t)m * N + n] = v;
            }
        }
}

// ---------------------------------------------------------------------------
// 3) RoPE + head reorder:  Y[b,s, h*64+2i(+1)] -> T[b,h,s, 2i(+1)]
//    mode 0: rope + 1/sqrt(dk) scale (Q), mode 1: rope (K), mode 2: copy (V)
// ---------------------------------------------------------------------------
__global__ void rope_reorder(const __bf16* __restrict__ Y,
                             __bf16* __restrict__ T, int mode) {
    int idx = blockIdx.x * blockDim.x + threadIdx.x;   // pair index, 2^21 total
    int i = idx & 31;
    int h = (idx >> 5) & (NHEADS - 1);
    int s = (idx >> 9) & (SEQ - 1);
    int b = idx >> 20;

    const __bf16* src = Y + ((size_t)(b * SEQ + s)) * D_MODEL + h * DK + 2 * i;
    float t1 = (float)src[0];
    float t2 = (float)src[1];
    float r1, r2;
    if (mode == 2) {
        r1 = t1; r2 = t2;
    } else {
        float fr  = __powf(10000.0f, -(float)i / 32.0f);
        float ang = (float)s * fr;
        float c = __cosf(ang), sn = __sinf(ang);
        r1 = t1 * c - t2 * sn;
        r2 = t1 * sn + t2 * c;
        if (mode == 0) { r1 *= 0.125f; r2 *= 0.125f; }  // 1/sqrt(64)
    }
    __bf16* dst = T + (((size_t)(b * NHEADS + h)) * SEQ + s) * DK + 2 * i;
    dst[0] = (__bf16)r1;
    dst[1] = (__bf16)r2;
}

// ---------------------------------------------------------------------------
// 4) Flash attention, causal. Grid: (S/128, B*H). 8 waves x 16 q-rows.
//    K tile arrives via TDM (overlaps the manual transposed-V load);
//    per 32-key iteration: 4 WMMA (QK^T) + online softmax + 4 WMMA (PV).
// ---------------------------------------------------------------------------
__global__ __launch_bounds__(256) void attn_wmma(const __bf16* __restrict__ Q,
                                                 const __bf16* __restrict__ Kv,
                                                 const __bf16* __restrict__ Vv,
                                                 __bf16* __restrict__ O) {
    __shared__ __bf16 Kl[32 * 72];        // 32 keys x 64 dk, rows padded to 72
    __shared__ __bf16 Vt[64 * 40];        // transposed: 64 dk x 32 keys (pad 40)
    __shared__ __bf16 Pl[8 * 16 * 40];    // per-wave P tile: 16 q x 32 keys

    const int tid  = threadIdx.x;
    const int lane = tid & 31;
    const int w    = tid >> 5;
    const int lhi  = lane >> 4;
    const int llo  = lane & 15;
    const int bh   = blockIdx.y;          // b*H + h
    const int q0   = blockIdx.x * 128;
    const int qrow_w = q0 + w * 16;

    const size_t head_off = (size_t)bh * SEQ * DK;
    const __bf16* Qh = Q  + head_off;
    const __bf16* Kh = Kv + head_off;
    const __bf16* Vh = Vv + head_off;
    const unsigned kl_off = (unsigned)(size_t)(void*)&Kl[0];

    // Q fragments for this wave (A operand, dk=64 -> two K-steps), scaled
    // already by 1/sqrt(dk) in rope_reorder.
    Frag qa[2];
    {
        const __bf16* p = Qh + (size_t)(qrow_w + llo) * DK + lhi * 8;
        qa[0].u4[0] = *(const uint4*)p;
        qa[0].u4[1] = *(const uint4*)(p + 16);
        qa[1].u4[0] = *(const uint4*)(p + 32);
        qa[1].u4[1] = *(const uint4*)(p + 48);
    }

    v8f zero = {};
    v8f oacc[4];
#pragma unroll
    for (int nt = 0; nt < 4; ++nt) oacc[nt] = zero;
    float mrow[8], lrow[8];
#pragma unroll
    for (int r = 0; r < 8; ++r) { mrow[r] = -1e30f; lrow[r] = 0.0f; }

    const int ktmax = q0 / 32 + 3;        // causal: keys <= q0+127
    for (int kt = 0; kt <= ktmax; ++kt) {
        const int ko = kt * 32;
        // ---- K tile via TDM: 32 rows x 64 bf16, pad 64 -> 72 per row ----
        if (w == 0)
            tdm_load_tile_2d(kl_off, &Kh[(size_t)ko * DK],
                             DK, 32, 64, 32, DK, /*pad 32DW*/4, /*amt 4DW*/3);
        // ---- V tile transposed, manual (overlaps the K DMA) ----
        {
            int row = tid >> 3;
            int c8  = (tid & 7) * 8;
            uint4 t = *(const uint4*)&Vh[(size_t)(ko + row) * DK + c8];
            const __bf16* te = (const __bf16*)&t;
#pragma unroll
            for (int j = 0; j < 8; ++j) Vt[(c8 + j) * 40 + row] = te[j];
        }
        if (w == 0) __builtin_amdgcn_s_wait_tensorcnt(0);
        __syncthreads();

        // ---- scores = Q @ K^T : 2 n-tiles x 2 contraction steps ----
        v8f sc[2] = {zero, zero};
#pragma unroll
        for (int nt = 0; nt < 2; ++nt) {
            Frag b0, b1;
            const __bf16* p = &Kl[(nt * 16 + llo) * 72 + lhi * 8];
            b0.u4[0] = *(const uint4*)p;
            b0.u4[1] = *(const uint4*)(p + 16);
            b1.u4[0] = *(const uint4*)(p + 32);
            b1.u4[1] = *(const uint4*)(p + 48);
            sc[nt] = wmma_bf16(qa[0], b0, sc[nt]);
            sc[nt] = wmma_bf16(qa[1], b1, sc[nt]);
        }

        // ---- causal mask (only near the diagonal) ----
        if (ko + 31 > qrow_w) {
#pragma unroll
            for (int nt = 0; nt < 2; ++nt)
#pragma unroll
                for (int r = 0; r < 8; ++r) {
                    int key = ko + nt * 16 + llo;
                    int qr  = qrow_w + r + lhi * 8;
                    if (key > qr) sc[nt][r] = -1e30f;
                }
        }

        // ---- online softmax update (row reductions across 16 lanes) ----
        float scl[8];
#pragma unroll
        for (int r = 0; r < 8; ++r) {
            float mv = fmaxf(sc[0][r], sc[1][r]);
#pragma unroll
            for (int d = 1; d < 16; d <<= 1) mv = fmaxf(mv, __shfl_xor(mv, d, 32));
            float mn = fmaxf(mrow[r], mv);
            scl[r] = __expf(mrow[r] - mn);
            float e0 = __expf(sc[0][r] - mn);
            float e1 = __expf(sc[1][r] - mn);
            sc[0][r] = e0;
            sc[1][r] = e1;
            float ps = e0 + e1;
#pragma unroll
            for (int d = 1; d < 16; d <<= 1) ps += __shfl_xor(ps, d, 32);
            lrow[r] = lrow[r] * scl[r] + ps;
            mrow[r] = mn;
        }
#pragma unroll
        for (int nt = 0; nt < 4; ++nt)
#pragma unroll
            for (int r = 0; r < 8; ++r) oacc[nt][r] *= scl[r];

        // ---- P: C-layout -> LDS -> A-fragment (per-wave region) ----
        __bf16* Pw = &Pl[w * 16 * 40];
#pragma unroll
        for (int nt = 0; nt < 2; ++nt)
#pragma unroll
            for (int r = 0; r < 8; ++r)
                Pw[(r + lhi * 8) * 40 + nt * 16 + llo] = (__bf16)sc[nt][r];
        Frag pa;
        {
            const __bf16* p = &Pw[llo * 40 + lhi * 8];
            pa.u4[0] = *(const uint4*)p;
            pa.u4[1] = *(const uint4*)(p + 16);
        }

        // ---- O += P @ V : 4 dk n-tiles, contraction = 32 keys (1 step) ----
#pragma unroll
        for (int nt = 0; nt < 4; ++nt) {
            Frag vb;
            const __bf16* p = &Vt[(nt * 16 + llo) * 40 + lhi * 8];
            vb.u4[0] = *(const uint4*)p;
            vb.u4[1] = *(const uint4*)(p + 16);
            oacc[nt] = wmma_bf16(pa, vb, oacc[nt]);
        }
        __syncthreads();
    }

    // ---- normalize and store context as [b, s, h*64+d] bf16 ----
    const int b = bh >> 4, h = bh & 15;
#pragma unroll
    for (int r = 0; r < 8; ++r) lrow[r] = 1.0f / lrow[r];
#pragma unroll
    for (int nt = 0; nt < 4; ++nt)
#pragma unroll
        for (int r = 0; r < 8; ++r) {
            int srow = qrow_w + r + lhi * 8;
            size_t off = ((size_t)(b * SEQ + srow)) * D_MODEL + h * DK + nt * 16 + llo;
            O[off] = (__bf16)(oacc[nt][r] * lrow[r]);
        }
}

// ---------------------------------------------------------------------------
// Host-side launch. Workspace layout (bytes), total 64 MiB required:
//   [ 0, 8M)  x_bf        (and later reused for Octx)
//   [ 8,16M)  wq/wk/wv/wo bf16 (2 MiB each)
//   [16,40M)  yq/yk/yv    (8 MiB each)
//   [40,64M)  Q/K/V       (8 MiB each)
// ---------------------------------------------------------------------------
extern "C" void kernel_launch(void* const* d_in, const int* in_sizes, int n_in,
                              void* d_out, int out_size, void* d_ws, size_t ws_size,
                              hipStream_t stream) {
    const float* x  = (const float*)d_in[0];
    const float* wq = (const float*)d_in[1];
    const float* wk = (const float*)d_in[2];
    const float* wv = (const float*)d_in[3];
    const float* wo = (const float*)d_in[4];
    float* out = (float*)d_out;

    const size_t MB = 1024 * 1024;
    char* ws = (char*)d_ws;
    __bf16* x_bf  = (__bf16*)(ws + 0 * MB);
    __bf16* wq_bf = (__bf16*)(ws + 8 * MB);
    __bf16* wk_bf = (__bf16*)(ws + 10 * MB);
    __bf16* wv_bf = (__bf16*)(ws + 12 * MB);
    __bf16* wo_bf = (__bf16*)(ws + 14 * MB);
    __bf16* yq    = (__bf16*)(ws + 16 * MB);
    __bf16* yk    = (__bf16*)(ws + 24 * MB);
    __bf16* yv    = (__bf16*)(ws + 32 * MB);
    __bf16* Qb    = (__bf16*)(ws + 40 * MB);
    __bf16* Kb    = (__bf16*)(ws + 48 * MB);
    __bf16* Vb    = (__bf16*)(ws + 56 * MB);
    __bf16* Octx  = (__bf16*)(ws + 0 * MB);   // reuses x_bf region

    const int NX = MTOT * D_MODEL;            // 4,194,304
    const int NW = D_MODEL * D_MODEL;         // 1,048,576
    cvt_f32_bf16<<<NX / 256, 256, 0, stream>>>(x,  x_bf,  NX);
    cvt_f32_bf16<<<NW / 256, 256, 0, stream>>>(wq, wq_bf, NW);
    cvt_f32_bf16<<<NW / 256, 256, 0, stream>>>(wk, wk_bf, NW);
    cvt_f32_bf16<<<NW / 256, 256, 0, stream>>>(wv, wv_bf, NW);
    cvt_f32_bf16<<<NW / 256, 256, 0, stream>>>(wo, wo_bf, NW);

    dim3 ggrid(D_MODEL / 128, MTOT / 128);    // (8, 32)
    gemm_wmma<true><<<ggrid, 256, 0, stream>>>(x_bf, wq_bf, yq, MTOT, D_MODEL, D_MODEL);
    gemm_wmma<true><<<ggrid, 256, 0, stream>>>(x_bf, wk_bf, yk, MTOT, D_MODEL, D_MODEL);
    gemm_wmma<true><<<ggrid, 256, 0, stream>>>(x_bf, wv_bf, yv, MTOT, D_MODEL, D_MODEL);

    const int NPAIR = BATCH * SEQ * NHEADS * (DK / 2);   // 2,097,152
    rope_reorder<<<NPAIR / 256, 256, 0, stream>>>(yq, Qb, 0);
    rope_reorder<<<NPAIR / 256, 256, 0, stream>>>(yk, Kb, 1);
    rope_reorder<<<NPAIR / 256, 256, 0, stream>>>(yv, Vb, 2);

    dim3 agrid(SEQ / 128, BATCH * NHEADS);    // (16, 32)
    attn_wmma<<<agrid, 256, 0, stream>>>(Qb, Kb, Vb, Octx);

    gemm_wmma<false><<<ggrid, 256, 0, stream>>>(Octx, wo_bf, out, MTOT, D_MODEL, D_MODEL);
}
